// SoftSOM_44968307589672
// MI455X (gfx1250) — matrix operand order repaired and checked by reading, not compiled, not run on hardware
//
#include <hip/hip_runtime.h>
#include <hip/hip_bf16.h>
#include <math.h>

typedef __bf16 bf16;
typedef __attribute__((ext_vector_type(16))) __bf16 v16bf;
typedef __attribute__((ext_vector_type(8)))  __bf16 v8bf;
typedef __attribute__((ext_vector_type(8)))  float  v8f;
typedef __attribute__((ext_vector_type(4)))  float  v4f;

#define NPTS  32768
#define DIM   512
#define MPROT 4096

__device__ __forceinline__ v8f wmma_bf16(v16bf a, v16bf b, v8f c) {
  // 8 args: (neg_a, A, neg_b, B, c_mod, C, reuse_a, reuse_b)
  return __builtin_amdgcn_wmma_f32_16x16x32_bf16(false, a, false, b, (short)0, c,
                                                 false, false);
}

__device__ __forceinline__ v16bf combine(v8bf s0, v8bf s1) {
  return __builtin_shufflevector(s0, s1, 0, 1, 2, 3, 4, 5, 6, 7,
                                 8, 9, 10, 11, 12, 13, 14, 15);
}

// Load a 16-element bf16 fragment from two 16-byte segments 16 elements apart.
__device__ __forceinline__ v16bf ldfrag(const bf16* ptr) {
  return combine(*(const v8bf*)ptr, *(const v8bf*)(ptr + 16));
}

// Split 16 fp32 values into bf16 hi + bf16 residual-lo fragments (3xbf16 trick).
__device__ __forceinline__ void split16(const float* f, v16bf& h, v16bf& l) {
#pragma unroll
  for (int i = 0; i < 16; ++i) {
    bf16 hh = (bf16)f[i];
    h[i] = hh;
    l[i] = (bf16)(f[i] - (float)hh);
  }
}

// ---------------------------------------------------------------------------
// Kernel 1: row squared norms of x [N,D] and prototypes [M,D].
// One wave32 per row, 16 floats per lane, shfl-xor reduction.
// ---------------------------------------------------------------------------
__global__ void __launch_bounds__(256) k_norms(const float* __restrict__ x,
                                               const float* __restrict__ p,
                                               float* __restrict__ xsq,
                                               float* __restrict__ psq) {
  const int wave = (blockIdx.x * 256 + threadIdx.x) >> 5;
  const int lane = threadIdx.x & 31;
  const float* row;
  float* out;
  if (wave < NPTS) {
    row = x + (size_t)wave * DIM;
    out = xsq + wave;
  } else {
    row = p + (size_t)(wave - NPTS) * DIM;
    out = psq + (wave - NPTS);
  }
  float s = 0.0f;
#pragma unroll
  for (int i = 0; i < DIM; i += 128) {
    v4f v = *(const v4f*)(row + i + lane * 4);
    s += v.x * v.x + v.y * v.y + v.z * v.z + v.w * v.w;
  }
#pragma unroll
  for (int off = 16; off > 0; off >>= 1) s += __shfl_xor(s, off, 32);
  if (lane == 0) *out = s;
}

// ---------------------------------------------------------------------------
// Kernel 1b: pre-split prototypes into bf16 hi/lo, row-major [M][D] (for the
// distance GEMM) and transposed [D][M] (for the blend GEMM). One-time 8 MB.
// ---------------------------------------------------------------------------
__global__ void __launch_bounds__(256) k_cvt(const float* __restrict__ p,
                                             bf16* __restrict__ pr_hi,
                                             bf16* __restrict__ pr_lo,
                                             bf16* __restrict__ pt_hi,
                                             bf16* __restrict__ pt_lo) {
  const int i = blockIdx.x * 256 + threadIdx.x;  // i < M*D
  const int m = i >> 9;
  const int d = i & (DIM - 1);
  const float f = p[i];
  const bf16 h = (bf16)f;
  const bf16 l = (bf16)(f - (float)h);
  pr_hi[i] = h;
  pr_lo[i] = l;
  pt_hi[(size_t)d * MPROT + m] = h;
  pt_lo[(size_t)d * MPROT + m] = l;
}

// ---------------------------------------------------------------------------
// Kernel 2: scores = -cdist(x,P)/temp via split-bf16 WMMA GEMM (K=512).
// Raw scores -> weights region; per-row online max / sum-exp -> workspace.
// Block: 256 threads = 8 waves; 16 rows x 4096 cols; 4-column-tile register
// blocking so A fragments amortize over 12 WMMAs. B comes pre-split from ws.
// ---------------------------------------------------------------------------
__global__ void __launch_bounds__(256) k_scores(
    const float* __restrict__ x, const bf16* __restrict__ pr_hi,
    const bf16* __restrict__ pr_lo, const float* __restrict__ temp_raw,
    const float* __restrict__ xsq, const float* __restrict__ psq,
    float* __restrict__ wout, float* __restrict__ rmax,
    float* __restrict__ rinv) {
  __shared__ __attribute__((aligned(16))) bf16 a_hi[16][DIM];  // 16 KB
  __shared__ __attribute__((aligned(16))) bf16 a_lo[16][DIM];  // 16 KB
  __shared__ float pm[16][8];
  __shared__ float pl[16][8];

  const int tid = threadIdx.x;
  const int rowbase = blockIdx.x * 16;

  // Stage x tile as bf16 hi/lo in LDS (coalesced f32 reads).
  for (int i = tid; i < 16 * DIM; i += 256) {
    const int r = i >> 9;
    const int kk = i & (DIM - 1);
    const float f = x[(size_t)(rowbase + r) * DIM + kk];
    const bf16 h = (bf16)f;
    a_hi[r][kk] = h;
    a_lo[r][kk] = (bf16)(f - (float)h);
  }
  __syncthreads();

  const int wave = tid >> 5;
  const int lane = tid & 31;
  const int half = lane >> 4;
  const int ln = lane & 15;

  const float tr = temp_raw[0];
  const float temp = 1.0f / (1.0f + __expf(-tr)) * 0.999f + 0.001f;
  const float inv_t = 1.0f / temp;

  float xs[8];
#pragma unroll
  for (int v = 0; v < 8; ++v) xs[v] = xsq[rowbase + v + half * 8];

  float run_m[8], run_l[8];
#pragma unroll
  for (int v = 0; v < 8; ++v) {
    run_m[v] = -3.0e38f;
    run_l[v] = 0.0f;
  }

  const bf16* ah = &a_hi[ln][half * 8];
  const bf16* al = &a_lo[ln][half * 8];

  for (int tg = 0; tg < 8; ++tg) {
    int col[4];
    const bf16 *bh[4], *bl[4];
    v8f acc[4];
#pragma unroll
    for (int j = 0; j < 4; ++j) {
      col[j] = wave * 512 + (tg * 4 + j) * 16 + ln;
      bh[j] = pr_hi + (size_t)col[j] * DIM + half * 8;
      bl[j] = pr_lo + (size_t)col[j] * DIM + half * 8;
      acc[j] = (v8f){0, 0, 0, 0, 0, 0, 0, 0};
    }

    for (int kc = 0; kc < 16; ++kc) {
      const int k = kc * 32;
      // A fragment (x rows) from LDS, shared by the 4 column tiles.
      v16bf Ah = combine(*(const v8bf*)(ah + k), *(const v8bf*)(ah + k + 16));
      v16bf Al = combine(*(const v8bf*)(al + k), *(const v8bf*)(al + k + 16));
#pragma unroll
      for (int j = 0; j < 4; ++j) {
        v16bf Bh = ldfrag(bh[j] + k);
        v16bf Bl = ldfrag(bl[j] + k);
        acc[j] = wmma_bf16(Ah, Bh, acc[j]);
        acc[j] = wmma_bf16(Ah, Bl, acc[j]);
        acc[j] = wmma_bf16(Al, Bh, acc[j]);
      }
    }

    // dist -> score; emit raw scores; per-lane online softmax stats.
#pragma unroll
    for (int j = 0; j < 4; ++j) {
      const float ps = psq[col[j]];
#pragma unroll
      for (int v = 0; v < 8; ++v) {
        const float d2 = xs[v] + ps - 2.0f * acc[j][v];
        const float d = __builtin_amdgcn_sqrtf(fmaxf(d2, 0.0f));
        const float s = -d * inv_t;
        wout[(size_t)(rowbase + v + half * 8) * MPROT + col[j]] = s;
        const float nm = fmaxf(run_m[v], s);
        run_l[v] = run_l[v] * __expf(run_m[v] - nm) + __expf(s - nm);
        run_m[v] = nm;
      }
    }
  }

  // Cross-lane log-sum-exp merge within each 16-lane half.
#pragma unroll
  for (int v = 0; v < 8; ++v) {
#pragma unroll
    for (int off = 1; off < 16; off <<= 1) {
      const float om = __shfl_xor(run_m[v], off, 32);
      const float ol = __shfl_xor(run_l[v], off, 32);
      const float nm = fmaxf(run_m[v], om);
      run_l[v] = run_l[v] * __expf(run_m[v] - nm) + ol * __expf(om - nm);
      run_m[v] = nm;
    }
  }

  if (ln == 0) {
#pragma unroll
    for (int v = 0; v < 8; ++v) {
      pm[v + half * 8][wave] = run_m[v];
      pl[v + half * 8][wave] = run_l[v];
    }
  }
  __syncthreads();
  if (tid < 16) {
    float m = -3.0e38f;
    for (int w = 0; w < 8; ++w) m = fmaxf(m, pm[tid][w]);
    float l = 0.0f;
    for (int w = 0; w < 8; ++w) l += pl[tid][w] * __expf(pm[tid][w] - m);
    rmax[rowbase + tid] = m;
    rinv[rowbase + tid] = 1.0f / l;
  }
}

// ---------------------------------------------------------------------------
// Kernel 3: in-place softmax normalization of the weights matrix (v4 wide).
// ---------------------------------------------------------------------------
__global__ void __launch_bounds__(256) k_norm_weights(
    float* __restrict__ w, const float* __restrict__ rmax,
    const float* __restrict__ rinv) {
  const size_t stride = (size_t)gridDim.x * 256;
  const size_t total4 = (size_t)NPTS * MPROT / 4;
  v4f* w4 = (v4f*)w;
  for (size_t i = (size_t)blockIdx.x * 256 + threadIdx.x; i < total4;
       i += stride) {
    const int row = (int)(i >> 10);  // (i*4) >> 12
    const float m = rmax[row];
    const float iv = rinv[row];
    v4f v = w4[i];
    v.x = __expf(v.x - m) * iv;
    v.y = __expf(v.y - m) * iv;
    v.z = __expf(v.z - m) * iv;
    v.w = __expf(v.w - m) * iv;
    w4[i] = v;
  }
}

// ---------------------------------------------------------------------------
// Kernel 4: blended = weights [N,M] @ prototypes [M,D], split-bf16 WMMA.
// Block: 256 threads = 8 waves; 16 rows x 512 cols. B fragments load
// directly from the transposed pre-split copy (L2-resident); no LDS at all.
// ---------------------------------------------------------------------------
__global__ void __launch_bounds__(256) k_blend(const bf16* __restrict__ pt_hi,
                                               const bf16* __restrict__ pt_lo,
                                               const float* __restrict__ w,
                                               float* __restrict__ out) {
  const int tid = threadIdx.x;
  const int rowbase = blockIdx.x * 16;
  const int wave = tid >> 5;
  const int lane = tid & 31;
  const int half = lane >> 4;
  const int ln = lane & 15;

  v8f acc[4];
  const bf16 *bh[4], *bl[4];
#pragma unroll
  for (int ct = 0; ct < 4; ++ct) {
    acc[ct] = (v8f){0, 0, 0, 0, 0, 0, 0, 0};
    const int bcol = wave * 64 + ct * 16 + ln;
    bh[ct] = pt_hi + (size_t)bcol * MPROT + half * 8;
    bl[ct] = pt_lo + (size_t)bcol * MPROT + half * 8;
  }

  const float* arow = w + (size_t)(rowbase + ln) * MPROT + half * 8;

  for (int kc = 0; kc < 128; ++kc) {
    const int k = kc * 32;
    // A fragment: 16 fp32 weights per lane (two 32B segments), split here;
    // cost amortized over the 12 WMMAs below.
    v4f a0 = *(const v4f*)(arow + k);
    v4f a1 = *(const v4f*)(arow + k + 4);
    v4f a2 = *(const v4f*)(arow + k + 16);
    v4f a3 = *(const v4f*)(arow + k + 20);
    float fa[16] = {a0.x, a0.y, a0.z, a0.w, a1.x, a1.y, a1.z, a1.w,
                    a2.x, a2.y, a2.z, a2.w, a3.x, a3.y, a3.z, a3.w};
    v16bf Ah, Al;
    split16(fa, Ah, Al);

#pragma unroll
    for (int ct = 0; ct < 4; ++ct) {
      v16bf Bh = ldfrag(bh[ct] + k);
      v16bf Bl = ldfrag(bl[ct] + k);
      acc[ct] = wmma_bf16(Ah, Bh, acc[ct]);
      acc[ct] = wmma_bf16(Ah, Bl, acc[ct]);
      acc[ct] = wmma_bf16(Al, Bh, acc[ct]);
    }
  }

#pragma unroll
  for (int ct = 0; ct < 4; ++ct) {
    const int col = wave * 64 + ct * 16 + ln;
#pragma unroll
    for (int v = 0; v < 8; ++v) {
      out[(size_t)(rowbase + v + half * 8) * DIM + col] = acc[ct][v];
    }
  }
}

// ---------------------------------------------------------------------------
extern "C" void kernel_launch(void* const* d_in, const int* in_sizes, int n_in,
                              void* d_out, int out_size, void* d_ws,
                              size_t ws_size, hipStream_t stream) {
  (void)in_sizes; (void)n_in; (void)out_size; (void)ws_size;
  const float* x = (const float*)d_in[0];
  const float* p = (const float*)d_in[1];
  const float* tr = (const float*)d_in[2];

  float* blended = (float*)d_out;                 // [N, D]
  float* weights = blended + (size_t)NPTS * DIM;  // [N, M]

  // Workspace layout (~17.2 MB total):
  //   [0)          xsq   : N floats
  //   [N)          psq   : M floats
  //   [N+M)        rmax  : N floats
  //   [2N+M)       rinv  : N floats   -> stats end at 409,600 B
  //   then bf16 pr_hi/pr_lo [M][D] and pt_hi/pt_lo [D][M], 4 MB each.
  float* wsf = (float*)d_ws;
  float* xsq = wsf;
  float* psq = wsf + NPTS;
  float* rmax = wsf + NPTS + MPROT;
  float* rinv = wsf + 2 * NPTS + MPROT;
  bf16* pr_hi = (bf16*)((char*)d_ws + 409600);
  bf16* pr_lo = pr_hi + (size_t)MPROT * DIM;
  bf16* pt_hi = pr_lo + (size_t)MPROT * DIM;
  bf16* pt_lo = pt_hi + (size_t)MPROT * DIM;

  k_norms<<<(NPTS + MPROT) / 8, 256, 0, stream>>>(x, p, xsq, psq);
  k_cvt<<<(MPROT * DIM) / 256, 256, 0, stream>>>(p, pr_hi, pr_lo, pt_hi,
                                                 pt_lo);
  k_scores<<<NPTS / 16, 256, 0, stream>>>(x, pr_hi, pr_lo, tr, xsq, psq,
                                          weights, rmax, rinv);
  k_norm_weights<<<32768, 256, 0, stream>>>(weights, rmax, rinv);
  k_blend<<<NPTS / 16, 256, 0, stream>>>(pt_hi, pt_lo, weights, blended);
}
